// TransformerBlock_23570780520791
// MI455X (gfx1250) — compile-verified, matrix operand order checked by reading, not run on hardware
//
#include <hip/hip_runtime.h>

// ---------------------------------------------------------------------------
// Point-Transformer block for MI455X (gfx1250, wave32, WMMA bf16 16x16x32).
// All heavy matmuls go through v_wmma_f32_16x16x32_bf16 with f32 accumulate.
// B fragments are double-buffered in VGPRs so global loads for tile nt+1
// overlap the WMMA block of tile nt (no s_wait_loadcnt 0 bubble per K-step).
// ---------------------------------------------------------------------------

typedef __attribute__((ext_vector_type(16))) __bf16   v16bf;
typedef __attribute__((ext_vector_type(8)))  float    v8f;
typedef __attribute__((ext_vector_type(4)))  unsigned int u32x4;

constexpr int cB = 4, cN = 4096, cK = 16, cD = 256;

__device__ inline unsigned short f2bf(float f) {
  unsigned int u = __float_as_uint(f);
  u += 0x7FFFu + ((u >> 16) & 1u);             // round-to-nearest-even
  return (unsigned short)(u >> 16);
}
__device__ inline float bf2f(unsigned short h) {
  return __uint_as_float(((unsigned int)h) << 16);
}
__device__ inline unsigned int pack2(unsigned short a, unsigned short b) {
  return (unsigned int)a | ((unsigned int)b << 16);
}

union FragU { u32x4 u[2]; v16bf v; };

// A fragment (16x32 bf16) from a row-major [16][256] bf16 LDS buffer.
// Lanes 0-15: row=lane, K = kk+{0..7, 16..23}; lanes 16-31: row=lane-16,
// K = kk+{8..15, 24..31}.  Two 16B ds loads per lane.
__device__ inline v16bf afrag_lds(const unsigned short* sA, int kk, int lane) {
  int r  = lane & 15;
  int ko = (lane >> 4) << 3;
  FragU f;
  f.u[0] = *(const u32x4*)(sA + r * cD + kk + ko);
  f.u[1] = *(const u32x4*)(sA + r * cD + kk + ko + 16);
  return f.v;
}

// B fragment (32x16 bf16) from a global W^T (N x K, row-major, K=256) buffer.
// Lane = output column; per-lane K packing mirrors the A layout.
__device__ inline v16bf bfrag_gl(const unsigned short* WT, int kk, int nn, int lane) {
  int c  = lane & 15;
  int ko = (lane >> 4) << 3;
  const unsigned short* p = WT + (nn + c) * cD + kk + ko;
  FragU f;
  f.u[0] = *(const u32x4*)p;
  f.u[1] = *(const u32x4*)(p + 16);
  return f.v;
}

// Load all 8 B fragments (full K=256) of one 16-wide N-tile into registers.
__device__ inline void loadBfrags(v16bf* dst, const unsigned short* WT, int nn, int lane) {
#pragma unroll
  for (int kt = 0; kt < 8; ++kt) dst[kt] = bfrag_gl(WT, kt * 32, nn, lane);
}

// One wave computes OUT(16x256) = A_lds(16x256) @ W^T, K=256.
// A register-resident, B double-buffered; epilogue is a functor taking
// (nt, acc) with the standard 16x16 f32 C layout.
template <class EPI>
__device__ inline void wave_mm_core(const unsigned short* __restrict__ sA,
                                    const unsigned short* __restrict__ WT,
                                    int lane, EPI epi) {
  v16bf af[8];
#pragma unroll
  for (int kt = 0; kt < 8; ++kt) af[kt] = afrag_lds(sA, kt * 32, lane);

  v16bf b0[8], b1[8];
  loadBfrags(b0, WT, 0, lane);
  for (int ntp = 0; ntp < 8; ++ntp) {
    loadBfrags(b1, WT, (2 * ntp + 1) * 16, lane);     // prefetch odd tile
    v8f acc = {};
#pragma unroll
    for (int kt = 0; kt < 8; ++kt)
      acc = __builtin_amdgcn_wmma_f32_16x16x32_bf16(false, af[kt], false, b0[kt],
                                                    (short)0, acc, false, false);
    epi(2 * ntp, acc);
    if (ntp < 7) loadBfrags(b0, WT, (2 * ntp + 2) * 16, lane);  // prefetch next even
    v8f acc2 = {};
#pragma unroll
    for (int kt = 0; kt < 8; ++kt)
      acc2 = __builtin_amdgcn_wmma_f32_16x16x32_bf16(false, af[kt], false, b1[kt],
                                                     (short)0, acc2, false, false);
    epi(2 * ntp + 1, acc2);
  }
}

// ---------------------------------------------------------------------------
// Weight prep kernels
// ---------------------------------------------------------------------------
__global__ void __launch_bounds__(256)
k_transpose_bf16(const float* __restrict__ src, unsigned short* __restrict__ dst,
                 int rows, int cols) {
  int i = blockIdx.x * 256 + threadIdx.x;
  if (i >= rows * cols) return;
  int r = i / cols, c = i % cols;
  dst[c * rows + r] = f2bf(src[i]);
}

__global__ void __launch_bounds__(256)
k_convert_bf16(const float* __restrict__ src, unsigned short* __restrict__ dst, int n) {
  int i = blockIdx.x * 256 + threadIdx.x;
  if (i < n) dst[i] = f2bf(src[i]);
}

// ---------------------------------------------------------------------------
// kNN: one query point per thread, xyz tiled through LDS, register top-16.
// Downstream ops are permutation-invariant over K so ordering is irrelevant.
// ---------------------------------------------------------------------------
__global__ void __launch_bounds__(128)
k_knn(const float* __restrict__ xyz, int* __restrict__ idx_out) {
  __shared__ float sx[256 * 3];
  int tid  = threadIdx.x;
  int nblk = cN / 128;
  int b = blockIdx.x / nblk;
  int n = (blockIdx.x % nblk) * 128 + tid;
  const float* base = xyz + b * cN * 3;
  float qx = base[n * 3 + 0], qy = base[n * 3 + 1], qz = base[n * 3 + 2];

  float bd[16];
  int   bi[16];
#pragma unroll
  for (int i = 0; i < 16; ++i) { bd[i] = 3.0e38f; bi[i] = 0; }
  float wmax = 3.0e38f;
  int   wpos = 0;

  for (int t = 0; t < cN; t += 256) {
    __syncthreads();
    for (int i = tid; i < 768; i += 128) sx[i] = base[t * 3 + i];
    __syncthreads();
    for (int j = 0; j < 256; ++j) {
      float dx = qx - sx[j * 3 + 0];
      float dy = qy - sx[j * 3 + 1];
      float dz = qz - sx[j * 3 + 2];
      float d2 = fmaf(dx, dx, fmaf(dy, dy, dz * dz));
      if (d2 < wmax) {
#pragma unroll
        for (int q = 0; q < 16; ++q)
          if (q == wpos) { bd[q] = d2; bi[q] = t + j; }
        float m = bd[0]; int p = 0;
#pragma unroll
        for (int q = 1; q < 16; ++q)
          if (bd[q] > m) { m = bd[q]; p = q; }
        wmax = m; wpos = p;
      }
    }
  }
  int* o = idx_out + (b * cN + n) * cK;
#pragma unroll
  for (int i = 0; i < 16; ++i) o[i] = bi[i];
}

// ---------------------------------------------------------------------------
// Row-block GEMM: one wave -> 16 rows x 256 cols, K=256.
// OUTMODE: 0 = f32 store, 1 = f32 accumulate, 2 = bf16 store (branch-free).
// ---------------------------------------------------------------------------
template <int OUTMODE, bool HASBIAS>
__global__ void __launch_bounds__(32)
k_gemm16(const unsigned short* __restrict__ A,
         const unsigned short* __restrict__ WT,
         const float* __restrict__ bias,
         float* __restrict__ outF, unsigned short* __restrict__ outB) {
  __shared__ __align__(16) unsigned short sA[16 * cD];
  int lane = threadIdx.x;
  int m0 = blockIdx.x * 16;
#pragma unroll
  for (int i = 0; i < 16; ++i) {
    int chunk = i * 32 + lane;               // 512 chunks of 8 bf16
    int r = chunk >> 5, c8 = (chunk & 31) * 8;
    *(u32x4*)(sA + r * cD + c8) = *(const u32x4*)(A + (m0 + r) * cD + c8);
  }
  __syncthreads();
  wave_mm_core(sA, WT, lane, [&](int nt, v8f acc) {
    int col = nt * 16 + (lane & 15);
    int mb  = (lane >> 4) * 8;
    float bv = HASBIAS ? bias[col] : 0.0f;
#pragma unroll
    for (int r = 0; r < 8; ++r) {
      float x = acc[r] + bv;
      int off = (m0 + mb + r) * cD + col;
      if (OUTMODE == 0)      outF[off] = x;
      else if (OUTMODE == 1) outF[off] += x;
      else                   outB[off] = f2bf(x);
    }
  });
}

// ---------------------------------------------------------------------------
// Fused per-point attention: one wave per (b,n) point, all 16x256 tiles in LDS.
// ---------------------------------------------------------------------------
__global__ void __launch_bounds__(32)
k_attn(const float* __restrict__ xyz, const int* __restrict__ knn_idx,
       const unsigned short* __restrict__ f_bf, const float* __restrict__ q_f32,
       const float* __restrict__ dw1, const float* __restrict__ db1,
       const unsigned short* __restrict__ dw2T, const float* __restrict__ db2,
       const unsigned short* __restrict__ wkT, const unsigned short* __restrict__ wvT,
       const unsigned short* __restrict__ g1T, const float* __restrict__ gb1,
       const unsigned short* __restrict__ g2T, const float* __restrict__ gb2,
       unsigned short* __restrict__ res_bf) {
  __shared__ __align__(16) unsigned short sKF[16 * cD];  // knn_f, reused as h2
  __shared__ __align__(16) unsigned short sH1[16 * cD];  // h1, reused as a(bf16)
  __shared__ __align__(16) unsigned short sPE[16 * cD];  // pos_enc (bf16)
  __shared__ float sV[16 * cD];                          // v (f32)
  __shared__ float sG[16 * cD];                          // gamma output (f32)
  __shared__ float sRel[64];
  __shared__ int   sIdx[16];

  int lane = threadIdx.x;
  int pt = blockIdx.x;            // 0 .. B*N-1
  int b  = pt / cN;
  int n  = pt % cN;
  const float* xb = xyz + b * cN * 3;

  if (lane < 16) {
    int id = knn_idx[pt * cK + lane];
    sIdx[lane] = id;
    sRel[lane * 4 + 0] = xb[n * 3 + 0] - xb[id * 3 + 0];
    sRel[lane * 4 + 1] = xb[n * 3 + 1] - xb[id * 3 + 1];
    sRel[lane * 4 + 2] = xb[n * 3 + 2] - xb[id * 3 + 2];
  }
  __syncthreads();

  // ---- positional MLP layer 1 (K=3 -> pure VALU): h1 = relu(rel@W1+b1) ----
  {
    int c0 = lane * 8;
    float w0[8], w1[8], w2[8], bb[8];
#pragma unroll
    for (int i = 0; i < 8; ++i) {
      w0[i] = dw1[0 * cD + c0 + i];
      w1[i] = dw1[1 * cD + c0 + i];
      w2[i] = dw1[2 * cD + c0 + i];
      bb[i] = db1[c0 + i];
    }
    for (int k = 0; k < 16; ++k) {
      float rx = sRel[k * 4 + 0], ry = sRel[k * 4 + 1], rz = sRel[k * 4 + 2];
      unsigned short h[8];
#pragma unroll
      for (int i = 0; i < 8; ++i) {
        float x = fmaf(rx, w0[i], fmaf(ry, w1[i], fmaf(rz, w2[i], bb[i])));
        h[i] = f2bf(fmaxf(x, 0.0f));
      }
      u32x4 u;
      u[0] = pack2(h[0], h[1]); u[1] = pack2(h[2], h[3]);
      u[2] = pack2(h[4], h[5]); u[3] = pack2(h[6], h[7]);
      *(u32x4*)(sH1 + k * cD + c0) = u;
    }
  }
  __syncthreads();

  // ---- pos_enc = h1 @ W2 + b2  (WMMA, bf16 out) ----
  wave_mm_core(sH1, dw2T, lane, [&](int nt, v8f acc) {
    int col = nt * 16 + (lane & 15);
    int mb  = (lane >> 4) * 8;
    float bv = db2[col];
#pragma unroll
    for (int r = 0; r < 8; ++r)
      sPE[(mb + r) * cD + col] = f2bf(acc[r] + bv);
  });
  __syncthreads();

  // ---- gather neighbor features: knn_f[k][:] = f[b, idx[k], :] ----
  for (int k = 0; k < 16; ++k) {
    int id = sIdx[k];
    *(u32x4*)(sKF + k * cD + lane * 8) =
        *(const u32x4*)(f_bf + (b * cN + id) * cD + lane * 8);
  }
  __syncthreads();

  // ---- v = knn_f @ wv  (WMMA, f32 out) ----
  wave_mm_core(sKF, wvT, lane, [&](int nt, v8f acc) {
    int col = nt * 16 + (lane & 15);
    int mb  = (lane >> 4) * 8;
#pragma unroll
    for (int r = 0; r < 8; ++r)
      sV[(mb + r) * cD + col] = acc[r];
  });
  __syncthreads();

  // ---- a = q - knn_f@wk + pos_enc  (WMMA + fused epilogue, bf16 out) ----
  const float* qrow = q_f32 + pt * cD;
  wave_mm_core(sKF, wkT, lane, [&](int nt, v8f acc) {
    int col = nt * 16 + (lane & 15);
    int mb  = (lane >> 4) * 8;
    float qv = qrow[col];
#pragma unroll
    for (int r = 0; r < 8; ++r) {
      float aval = qv - acc[r] + bf2f(sPE[(mb + r) * cD + col]);
      sH1[(mb + r) * cD + col] = f2bf(aval);
    }
  });
  __syncthreads();

  // ---- gamma MLP: h2 = relu(a@G1+b1) -> sKF (bf16) ----
  wave_mm_core(sH1, g1T, lane, [&](int nt, v8f acc) {
    int col = nt * 16 + (lane & 15);
    int mb  = (lane >> 4) * 8;
    float bv = gb1[col];
#pragma unroll
    for (int r = 0; r < 8; ++r)
      sKF[(mb + r) * cD + col] = f2bf(fmaxf(acc[r] + bv, 0.0f));
  });
  __syncthreads();

  // ---- g = h2 @ G2 + b2 -> sG (f32) ----
  wave_mm_core(sKF, g2T, lane, [&](int nt, v8f acc) {
    int col = nt * 16 + (lane & 15);
    int mb  = (lane >> 4) * 8;
    float bv = gb2[col];
#pragma unroll
    for (int r = 0; r < 8; ++r)
      sG[(mb + r) * cD + col] = acc[r] + bv;
  });
  __syncthreads();

  // ---- softmax over K (scale 1/sqrt(256)=1/16), reduce attn*(v+pe) ----
  {
    int c0 = lane * 8;
#pragma unroll
    for (int i = 0; i < 8; ++i) {
      int c = c0 + i;
      float vals[16];
      float m = -3.0e38f;
#pragma unroll
      for (int k = 0; k < 16; ++k) {
        float s = sG[k * cD + c] * 0.0625f;
        vals[k] = s;
        m = fmaxf(m, s);
      }
      float sum = 0.0f;
#pragma unroll
      for (int k = 0; k < 16; ++k) {
        float e = __expf(vals[k] - m);
        vals[k] = e;
        sum += e;
      }
      float inv = 1.0f / sum;
      float r = 0.0f;
#pragma unroll
      for (int k = 0; k < 16; ++k)
        r = fmaf(vals[k] * inv, sV[k * cD + c] + bf2f(sPE[k * cD + c]), r);
      res_bf[pt * cD + c] = f2bf(r);
    }
  }
}

// ---------------------------------------------------------------------------
// Launcher
// ---------------------------------------------------------------------------
extern "C" void kernel_launch(void* const* d_in, const int* in_sizes, int n_in,
                              void* d_out, int out_size, void* d_ws, size_t ws_size,
                              hipStream_t stream) {
  const float* xyz  = (const float*)d_in[0];
  const float* feat = (const float*)d_in[1];
  const float* dw1  = (const float*)d_in[2];
  const float* db1  = (const float*)d_in[3];
  const float* dw2  = (const float*)d_in[4];
  const float* db2  = (const float*)d_in[5];
  const float* fc1w = (const float*)d_in[6];
  const float* fc1b = (const float*)d_in[7];
  const float* wq   = (const float*)d_in[8];
  const float* wk   = (const float*)d_in[9];
  const float* wv   = (const float*)d_in[10];
  const float* g1   = (const float*)d_in[11];
  const float* gb1  = (const float*)d_in[12];
  const float* g2   = (const float*)d_in[13];
  const float* gb2  = (const float*)d_in[14];
  const float* fc2w = (const float*)d_in[15];
  const float* fc2b = (const float*)d_in[16];
  const float* scw  = (const float*)d_in[17];
  const float* scb  = (const float*)d_in[18];
  float* out = (float*)d_out;

  char* ws = (char*)d_ws;
  size_t off = 0;
  auto alloc = [&](size_t bytes) -> void* {
    void* p = ws + off;
    off += (bytes + 255) & ~(size_t)255;
    return p;
  };
  const size_t wbytes = (size_t)cD * cD * 2;
  unsigned short* wT_fc1 = (unsigned short*)alloc(wbytes);
  unsigned short* wT_wq  = (unsigned short*)alloc(wbytes);
  unsigned short* wT_wk  = (unsigned short*)alloc(wbytes);
  unsigned short* wT_wv  = (unsigned short*)alloc(wbytes);
  unsigned short* wT_dw2 = (unsigned short*)alloc(wbytes);
  unsigned short* wT_g1  = (unsigned short*)alloc(wbytes);
  unsigned short* wT_g2  = (unsigned short*)alloc(wbytes);
  unsigned short* wT_fc2 = (unsigned short*)alloc(wbytes);
  unsigned short* wT_sc  = (unsigned short*)alloc(wbytes);
  int*            idxbuf = (int*)alloc((size_t)cB * cN * cK * 4);
  unsigned short* featb  = (unsigned short*)alloc((size_t)cB * cN * cD * 2);
  unsigned short* f_bf   = (unsigned short*)alloc((size_t)cB * cN * cD * 2);
  float*          q_f32  = (float*)alloc((size_t)cB * cN * cD * 4);
  unsigned short* res_bf = (unsigned short*)alloc((size_t)cB * cN * cD * 2);

  const int M = cB * cN;            // 16384 rows
  const int wgrid = (cD * cD + 255) / 256;

  // 1) weight transposes (fp32 -> bf16, N-major) + feature conversion
  k_transpose_bf16<<<wgrid, 256, 0, stream>>>(fc1w, wT_fc1, cD, cD);
  k_transpose_bf16<<<wgrid, 256, 0, stream>>>(wq,   wT_wq,  cD, cD);
  k_transpose_bf16<<<wgrid, 256, 0, stream>>>(wk,   wT_wk,  cD, cD);
  k_transpose_bf16<<<wgrid, 256, 0, stream>>>(wv,   wT_wv,  cD, cD);
  k_transpose_bf16<<<wgrid, 256, 0, stream>>>(dw2,  wT_dw2, cD, cD);
  k_transpose_bf16<<<wgrid, 256, 0, stream>>>(g1,   wT_g1,  cD, cD);
  k_transpose_bf16<<<wgrid, 256, 0, stream>>>(g2,   wT_g2,  cD, cD);
  k_transpose_bf16<<<wgrid, 256, 0, stream>>>(fc2w, wT_fc2, cD, cD);
  k_transpose_bf16<<<wgrid, 256, 0, stream>>>(scw,  wT_sc,  cD, cD);
  k_convert_bf16<<<(M * cD + 255) / 256, 256, 0, stream>>>(feat, featb, M * cD);

  // 2) kNN indices
  k_knn<<<cB * (cN / 128), 128, 0, stream>>>(xyz, idxbuf);

  // 3) f = feat@fc1 + b1 (bf16), then q = f@wq (f32, no bias)
  k_gemm16<2, true ><<<M / 16, 32, 0, stream>>>(featb, wT_fc1, fc1b, nullptr, f_bf);
  k_gemm16<0, false><<<M / 16, 32, 0, stream>>>(f_bf,  wT_wq,  nullptr, q_f32, nullptr);

  // 4) fused per-point attention -> res (bf16)
  k_attn<<<M, 32, 0, stream>>>(xyz, idxbuf, f_bf, q_f32,
                               dw1, db1, wT_dw2, db2,
                               wT_wk, wT_wv, wT_g1, gb1, wT_g2, gb2, res_bf);

  // 5) out = res@fc2 + fc2_b, then out += shortcut@sc + sc_b
  k_gemm16<0, true><<<M / 16, 32, 0, stream>>>(res_bf, wT_fc2, fc2b, out, nullptr);
  k_gemm16<1, true><<<M / 16, 32, 0, stream>>>(featb,  wT_sc,  scb,  out, nullptr);
}
